// HierarchicalHPR_26620207301253
// MI455X (gfx1250) — matrix operand order repaired and checked by reading, not compiled, not run on hardware
//
// HierarchicalHPR for MI455X (gfx1250, wave32, WMMA).
// Persistent kernel: 128 WGs x 256 threads; each WG owns 16 batch rows and
// runs the full T=512 recurrence with LDS-resident f16 weights and
// v_wmma_f32_16x16x32_f16 for the GRU / output GEMMs.
#include <hip/hip_runtime.h>

typedef __attribute__((ext_vector_type(16))) _Float16 v16h;
typedef __attribute__((ext_vector_type(8)))  _Float16 v8h;
typedef __attribute__((ext_vector_type(8)))  float    v8f;

namespace {
constexpr int kB = 2048, kT = 512, kIN = 64, kHID = 128, kSD = 4, kOUT = 32, kTS = 5;
constexpr int kG3 = 384;          // 3*HID
constexpr int kKA = 224;          // A row: [inp 64 | x1_pred 4 | pad 28 | h 128]
constexpr int kKX = 96;           // K for input-side GEMM (68 padded to 96)
constexpr int kKH = 128;          // K for hidden-side GEMM
constexpr int kNO = 48;           // meas(4) + out(32) + pad(12)
constexpr int kMT = 16;           // batch rows per workgroup
constexpr int kThreads = 256;

// float offsets into the PRM (small parameters) LDS blob
constexpr int PO_L2W1 = 0, PO_L2B1 = 128, PO_L2W2 = 160, PO_L2B2 = 288;
constexpr int PO_H2 = 292, PO_EQ2 = 308, PO_ER2 = 312;
constexpr int PO_L1W1 = 320, PO_L1B1 = 832, PO_L1W2 = 896, PO_L1B2 = 1152;
constexpr int PO_H1 = 1156, PO_EQ1 = 1172, PO_ER1 = 1176;
constexpr int kPRM = 1184;

// dynamic LDS carve (bytes); every base is 16B aligned
constexpr int SM_WX  = 0;
constexpr int SM_WH  = SM_WX  + kG3 * kKX * 2;     // 73728
constexpr int SM_WO  = SM_WH  + kG3 * kKH * 2;     // 98304
constexpr int SM_A   = SM_WO  + kNO * kKH * 2;     // 12288
constexpr int SM_GX  = SM_A   + kMT * kKA * 2;     // 7168
constexpr int SM_GH  = SM_GX  + kMT * kG3 * 4;     // 24576
constexpr int SM_O2  = SM_GH  + kMT * kG3 * 4;     // 24576
constexpr int SM_HF  = SM_O2  + kMT * kNO * 4;     // 3072
constexpr int SM_BIH = SM_HF  + kMT * kHID * 4;    // 8192
constexpr int SM_BHH = SM_BIH + kG3 * 4;
constexpr int SM_B2  = SM_BHH + kG3 * 4;
constexpr int SM_PRM = SM_B2  + kNO * 4;
constexpr int SM_X1  = SM_PRM + kPRM * 4;
constexpr int SM_X2  = SM_X1  + kMT * 4 * 4;
constexpr int SM_X1P = SM_X2  + kMT * 4 * 4;
constexpr int SM_X2P = SM_X1P + kMT * 4 * 4;
constexpr int SM_P1  = SM_X2P + kMT * 4 * 4;
constexpr int SM_P2  = SM_P1  + kMT * 16 * 4;
constexpr int SM_P1P = SM_P2  + kMT * 16 * 4;
constexpr int SM_SCR = SM_P1P + kMT * 16 * 4;      // per-row 96-float scratch
constexpr int SM_HSC = SM_SCR + kMT * 96 * 4;      // per-row 64-float MLP hidden
constexpr int SM_TOTAL = SM_HSC + kMT * 64 * 4;    // ~268 KB < 320 KB/WGP
} // namespace

// ---- CDNA5 WMMA fragment loads (wave32; A and B share this code's contiguous k order) ----
// A (16x32 f16): lanes 0-15 -> M=lane, K {0..15}; lanes 16-31 -> K {16..31}
__device__ inline v16h lds_load_a(const _Float16* Arow, int kbase, int half) {
  const _Float16* p = Arow + kbase + half * 16;
  v8h lo = *(const v8h*)p;
  v8h hi = *(const v8h*)(p + 8);
  v16h r;
#pragma unroll
  for (int i = 0; i < 8; ++i) { r[i] = lo[i]; r[i + 8] = hi[i]; }
  return r;
}
// B (32x16 f16): lane n = lane&15, 16 contiguous K-halves (half-wave K split)
__device__ inline v16h lds_load_b(const _Float16* p) {
  v8h lo = *(const v8h*)p;
  v8h hi = *(const v8h*)(p + 8);
  v16h r;
#pragma unroll
  for (int i = 0; i < 8; ++i) { r[i] = lo[i]; r[i + 8] = hi[i]; }
  return r;
}
// D (16x16 f32): VGPR r -> M = r + 8*half, N = lane&15
__device__ inline void store_acc(float* G, int stride, const v8f& c, int nb, int half, int nn) {
#pragma unroll
  for (int r = 0; r < 8; ++r) G[(r + half * 8) * stride + nb + nn] = c[r];
}

// ---- 4x4 Kalman update, 16 lanes of one wave cooperate on one batch row ----
// scr: [0..15]=S  [16..31]=PH^T  [32..47]=K  [48..63]=(I-KH)  [64..67]=y
__device__ inline void kf4(int l, const float* Hm, const float* eR,
                           const float* xpred, const float* Pp, const float* meas,
                           float* xo, float* Po, float* scr) {
  const int i = l >> 2, j = l & 3;
  float* S  = scr;
  float* Hp = scr + 16;
  float* Kk = scr + 32;
  float* Tt = scr + 48;
  float* yv = scr + 64;
  // S = H P H^T + diag(R) + 1e-6 I
  float s = 0.f;
#pragma unroll
  for (int k = 0; k < 4; ++k) {
    float hik = Hm[i * 4 + k];
#pragma unroll
    for (int q = 0; q < 4; ++q) s += hik * Pp[k * 4 + q] * Hm[j * 4 + q];
  }
  if (i == j) s += eR[i] + 1e-6f;
  S[l] = s;
  // PH^T
  float hp = 0.f;
#pragma unroll
  for (int k = 0; k < 4; ++k) hp += Pp[i * 4 + k] * Hm[j * 4 + k];
  Hp[l] = hp;
  // y = meas - x_pred H^T
  if (l < 4) {
    float y = meas[l];
#pragma unroll
    for (int k = 0; k < 4; ++k) y -= Hm[l * 4 + k] * xpred[k];
    yv[l] = y;
  }
  // K = (PH^T) S^{-1}  via  X = S^{-1} (PH^T)^T, K = X^T  (lane 0, Gauss-Jordan)
  if (l == 0) {
    float Ac[16], Bx[16];
#pragma unroll
    for (int a = 0; a < 16; ++a) Ac[a] = S[a];
#pragma unroll
    for (int r = 0; r < 4; ++r)
#pragma unroll
      for (int c = 0; c < 4; ++c) Bx[r * 4 + c] = Hp[c * 4 + r];
#pragma unroll
    for (int p = 0; p < 4; ++p) {
      float piv = 1.f / Ac[p * 4 + p];
#pragma unroll
      for (int c = 0; c < 4; ++c) { Ac[p * 4 + c] *= piv; Bx[p * 4 + c] *= piv; }
#pragma unroll
      for (int r = 0; r < 4; ++r) if (r != p) {
        float f = Ac[r * 4 + p];
#pragma unroll
        for (int c = 0; c < 4; ++c) { Ac[r * 4 + c] -= f * Ac[p * 4 + c]; Bx[r * 4 + c] -= f * Bx[p * 4 + c]; }
      }
    }
#pragma unroll
    for (int r = 0; r < 4; ++r)
#pragma unroll
      for (int c = 0; c < 4; ++c) Kk[r * 4 + c] = Bx[c * 4 + r];
  }
  // x_post = x_pred + K y
  if (l < 4) {
    float xv = xpred[l];
#pragma unroll
    for (int k = 0; k < 4; ++k) xv += Kk[l * 4 + k] * yv[k];
    xo[l] = xv;
  }
  // (I - K H)
  float tt = (i == j) ? 1.f : 0.f;
#pragma unroll
  for (int k = 0; k < 4; ++k) tt -= Kk[i * 4 + k] * Hm[k * 4 + j];
  Tt[l] = tt;
  // P_post = (I - K H) P_pred
  float pn = 0.f;
#pragma unroll
  for (int k = 0; k < 4; ++k) pn += Tt[i * 4 + k] * Pp[k * 4 + j];
  Po[l] = pn;
}

__global__ __launch_bounds__(kThreads)
void hpr_kernel(const float* __restrict__ inp,
                const float* l2w1, const float* l2b1, const float* l2w2, const float* l2b2,
                const float* l2H, const float* l2lQ, const float* l2lR,
                const float* l1w1, const float* l1b1, const float* l1w2, const float* l1b2,
                const float* l1H, const float* l1lQ, const float* l1lR,
                const float* wih, const float* whh, const float* bih, const float* bhh,
                const float* measw, const float* measb, const float* outw, const float* outb,
                float* __restrict__ outp) {
  extern __shared__ char smem[];
  _Float16* Wx = (_Float16*)(smem + SM_WX);
  _Float16* Wh = (_Float16*)(smem + SM_WH);
  _Float16* Wo = (_Float16*)(smem + SM_WO);
  _Float16* Af = (_Float16*)(smem + SM_A);
  float* GX  = (float*)(smem + SM_GX);
  float* GH  = (float*)(smem + SM_GH);
  float* O2  = (float*)(smem + SM_O2);
  float* HF  = (float*)(smem + SM_HF);
  float* BIHl = (float*)(smem + SM_BIH);
  float* BHHl = (float*)(smem + SM_BHH);
  float* B2l  = (float*)(smem + SM_B2);
  float* PRM  = (float*)(smem + SM_PRM);
  float* X1  = (float*)(smem + SM_X1);
  float* X2  = (float*)(smem + SM_X2);
  float* X1P = (float*)(smem + SM_X1P);
  float* X2P = (float*)(smem + SM_X2P);
  float* P1  = (float*)(smem + SM_P1);
  float* P2  = (float*)(smem + SM_P2);
  float* P1P = (float*)(smem + SM_P1P);
  float* SCR = (float*)(smem + SM_SCR);
  float* HSC = (float*)(smem + SM_HSC);

  const int tid = threadIdx.x;
  const int row0 = blockIdx.x * kMT;

  // ---------------- one-time staging: weights -> LDS (f16), params, state ----
  for (int i = tid; i < kG3 * kKX; i += kThreads) {
    int n = i / kKX, k = i % kKX;
    Wx[i] = (k < kIN + kSD) ? (_Float16)wih[n * (kIN + kSD) + k] : (_Float16)0.f;
  }
  for (int i = tid; i < kG3 * kKH; i += kThreads) Wh[i] = (_Float16)whh[i];
  for (int i = tid; i < kNO * kKH; i += kThreads) {
    int n = i / kKH, k = i % kKH;
    _Float16 v = (_Float16)0.f;
    if (n < 4) v = (_Float16)measw[n * kKH + k];
    else if (n < 36) v = (_Float16)outw[(n - 4) * kKH + k];
    Wo[i] = v;
  }
  for (int i = tid; i < kG3; i += kThreads) { BIHl[i] = bih[i]; BHHl[i] = bhh[i]; }
  for (int i = tid; i < kNO; i += kThreads) {
    float v = 0.f;
    if (i < 4) v = measb[i]; else if (i < 36) v = outb[i - 4];
    B2l[i] = v;
  }
  for (int i = tid; i < 128; i += kThreads) PRM[PO_L2W1 + i] = l2w1[i];
  for (int i = tid; i < 32;  i += kThreads) PRM[PO_L2B1 + i] = l2b1[i];
  for (int i = tid; i < 128; i += kThreads) PRM[PO_L2W2 + i] = l2w2[i];
  for (int i = tid; i < 4;   i += kThreads) PRM[PO_L2B2 + i] = l2b2[i];
  for (int i = tid; i < 16;  i += kThreads) PRM[PO_H2 + i] = l2H[i];
  for (int i = tid; i < 512; i += kThreads) PRM[PO_L1W1 + i] = l1w1[i];
  for (int i = tid; i < 64;  i += kThreads) PRM[PO_L1B1 + i] = l1b1[i];
  for (int i = tid; i < 256; i += kThreads) PRM[PO_L1W2 + i] = l1w2[i];
  for (int i = tid; i < 4;   i += kThreads) PRM[PO_L1B2 + i] = l1b2[i];
  for (int i = tid; i < 16;  i += kThreads) PRM[PO_H1 + i] = l1H[i];
  if (tid < 4) {
    PRM[PO_EQ2 + tid] = __expf(l2lQ[tid]);
    PRM[PO_ER2 + tid] = __expf(l2lR[tid]);
    PRM[PO_EQ1 + tid] = __expf(l1lQ[tid]);
    PRM[PO_ER1 + tid] = __expf(l1lR[tid]);
  }
  for (int i = tid; i < kMT * kKA; i += kThreads) Af[i] = (_Float16)0.f;   // h=0, pads=0
  for (int i = tid; i < kMT * kHID; i += kThreads) HF[i] = 0.f;
  for (int i = tid; i < kMT * 4; i += kThreads) { X1[i] = 0.f; X2[i] = 0.f; }
  for (int i = tid; i < kMT * 16; i += kThreads) {
    int e = i & 15;
    float v = ((e >> 2) == (e & 3)) ? 1.f : 0.f;
    P1[i] = v; P2[i] = v;
  }
  __syncthreads();

  const int wid = tid >> 5, lane = tid & 31;
  const int half = lane >> 4, lm = lane & 15;
  const _Float16* Arow = Af + lm * kKA;

  // ---------------- sequential scan over T ----------------
  for (int t = 0; t < kT; ++t) {
    // ---- phase 1: stream inp_t into A (f16) + small scalar path -> x1_pred
    {
      int idx = tid << 2;                 // 1024 floats = 16 rows x 64
      int m = idx >> 6, c = idx & 63;
      const float* src = inp + ((size_t)(row0 + m) * kT + t) * kIN + c;
      float4 v = *(const float4*)src;
      _Float16* dst = Af + m * kKA + c;
      dst[0] = (_Float16)v.x; dst[1] = (_Float16)v.y;
      dst[2] = (_Float16)v.z; dst[3] = (_Float16)v.w;
      if (t + 1 < kT) __builtin_prefetch((const void*)(src + kIN), 0, 2); // global_prefetch_b8
    }
    {
      const int m = tid >> 4, l = tid & 15;     // 16 lanes of one wave per row
      float* scr = SCR + m * 96;
      if ((t % kTS) == 0) {                     // level-2 slow path
        for (int rep = 0; rep < 2; ++rep) {     // hidden(32) = tanh(W1 x2 + b1)
          int j = l + rep * 16;
          float a = PRM[PO_L2B1 + j];
#pragma unroll
          for (int s = 0; s < 4; ++s) a += PRM[PO_L2W1 + j * 4 + s] * X2[m * 4 + s];
          HSC[m * 64 + j] = tanhf(a);
        }
        if (l < 4) {                            // x2_pred = W2 hidden + b2
          float a = PRM[PO_L2B2 + l];
          for (int j = 0; j < 32; ++j) a += PRM[PO_L2W2 + l * 32 + j] * HSC[m * 64 + j];
          X2P[m * 4 + l] = a;
        }
        { int i = l >> 2, jj = l & 3;           // P2_pred = P2 + diag(expQ2)
          scr[80 + l] = P2[m * 16 + l] + ((i == jj) ? PRM[PO_EQ2 + i] : 0.f); }
        kf4(l, PRM + PO_H2, PRM + PO_ER2, X2P + m * 4, scr + 80, X1 + m * 4,
            X2 + m * 4, P2 + m * 16, scr);
      }
      // level-1 predict: x1_pred = W2 tanh(W1 [x1,x2] + b1) + b2
      for (int rep = 0; rep < 4; ++rep) {
        int j = l + rep * 16;
        float a = PRM[PO_L1B1 + j];
#pragma unroll
        for (int s = 0; s < 4; ++s) a += PRM[PO_L1W1 + j * 8 + s] * X1[m * 4 + s];
#pragma unroll
        for (int s = 0; s < 4; ++s) a += PRM[PO_L1W1 + j * 8 + 4 + s] * X2[m * 4 + s];
        HSC[m * 64 + j] = tanhf(a);
      }
      if (l < 4) {
        float a = PRM[PO_L1B2 + l];
        for (int j = 0; j < 64; ++j) a += PRM[PO_L1W2 + l * 64 + j] * HSC[m * 64 + j];
        X1P[m * 4 + l] = a;
        Af[m * kKA + kIN + l] = (_Float16)a;    // feed GEMM-X
      }
      { int i = l >> 2, jj = l & 3;             // P1_pred = P1 + diag(expQ1)
        P1P[m * 16 + l] = P1[m * 16 + l] + ((i == jj) ? PRM[PO_EQ1 + i] : 0.f); }
    }
    __syncthreads();

    // ---- phase 2: WMMA GEMMs -> gx (K=96) and gh (K=128), N=384
    {
      v8f ax0 = {}, ax1 = {}, ax2 = {};
#pragma unroll
      for (int s = 0; s < 3; ++s) {
        v16h a = lds_load_a(Arow, s * 32, half);
        const _Float16* wb = Wx + (size_t)(wid * 48 + lm) * kKX + s * 32 + half * 16;
        v16h b0 = lds_load_b(wb);
        v16h b1 = lds_load_b(wb + 16 * kKX);
        v16h b2 = lds_load_b(wb + 32 * kKX);
        ax0 = __builtin_amdgcn_wmma_f32_16x16x32_f16(false, a, false, b0, (short)0, ax0, false, false);
        ax1 = __builtin_amdgcn_wmma_f32_16x16x32_f16(false, a, false, b1, (short)0, ax1, false, false);
        ax2 = __builtin_amdgcn_wmma_f32_16x16x32_f16(false, a, false, b2, (short)0, ax2, false, false);
      }
      store_acc(GX, kG3, ax0, (wid * 3 + 0) * 16, half, lm);
      store_acc(GX, kG3, ax1, (wid * 3 + 1) * 16, half, lm);
      store_acc(GX, kG3, ax2, (wid * 3 + 2) * 16, half, lm);

      v8f ah0 = {}, ah1 = {}, ah2 = {};
#pragma unroll
      for (int s = 0; s < 4; ++s) {
        v16h a = lds_load_a(Arow, kKX + s * 32, half);   // h lives at cols 96..223
        const _Float16* wb = Wh + (size_t)(wid * 48 + lm) * kKH + s * 32 + half * 16;
        v16h b0 = lds_load_b(wb);
        v16h b1 = lds_load_b(wb + 16 * kKH);
        v16h b2 = lds_load_b(wb + 32 * kKH);
        ah0 = __builtin_amdgcn_wmma_f32_16x16x32_f16(false, a, false, b0, (short)0, ah0, false, false);
        ah1 = __builtin_amdgcn_wmma_f32_16x16x32_f16(false, a, false, b1, (short)0, ah1, false, false);
        ah2 = __builtin_amdgcn_wmma_f32_16x16x32_f16(false, a, false, b2, (short)0, ah2, false, false);
      }
      store_acc(GH, kG3, ah0, (wid * 3 + 0) * 16, half, lm);
      store_acc(GH, kG3, ah1, (wid * 3 + 1) * 16, half, lm);
      store_acc(GH, kG3, ah2, (wid * 3 + 2) * 16, half, lm);
    }
    __syncthreads();

    // ---- phase 3: GRU gates + h update (n-gate needs gx/gh separate)
    for (int i = tid; i < kMT * kHID; i += kThreads) {
      int m = i >> 7, j = i & 127;
      const float* gx = GX + m * kG3;
      const float* gh = GH + m * kG3;
      float xr = gx[j]       + BIHl[j];
      float hr = gh[j]       + BHHl[j];
      float xz = gx[128 + j] + BIHl[128 + j];
      float hz = gh[128 + j] + BHHl[128 + j];
      float xn = gx[256 + j] + BIHl[256 + j];
      float hn = gh[256 + j] + BHHl[256 + j];
      float r = 1.f / (1.f + __expf(-(xr + hr)));
      float z = 1.f / (1.f + __expf(-(xz + hz)));
      float n = tanhf(xn + r * hn);
      float ho = (1.f - z) * n + z * HF[i];
      HF[i] = ho;
      Af[m * kKA + kKX + j] = (_Float16)ho;     // hn feeds GEMM-O and next step
    }
    __syncthreads();

    // ---- phase 4: WMMA GEMM-O: [meas|out] = hn @ [meas_w;out_w]^T (N=48, K=128)
    if (wid < 3) {
      v8f acc = {};
#pragma unroll
      for (int s = 0; s < 4; ++s) {
        v16h a = lds_load_a(Arow, kKX + s * 32, half);
        const _Float16* wb = Wo + (size_t)(wid * 16 + lm) * kKH + s * 32 + half * 16;
        v16h b = lds_load_b(wb);
        acc = __builtin_amdgcn_wmma_f32_16x16x32_f16(false, a, false, b, (short)0, acc, false, false);
      }
      store_acc(O2, kNO, acc, wid * 16, half, lm);
    }
    __syncthreads();

    // ---- phase 5: stream out_t + level-1 Kalman update
    {
      int idx = tid << 1;                       // 512 outputs = 16 rows x 32
      int m = idx >> 5, o = idx & 31;
      float v0 = O2[m * kNO + 4 + o]     + B2l[4 + o];
      float v1 = O2[m * kNO + 4 + o + 1] + B2l[5 + o];
      float* dst = outp + ((size_t)(row0 + m) * kT + t) * kOUT + o;
      dst[0] = v0; dst[1] = v1;
    }
    {
      const int m = tid >> 4, l = tid & 15;
      float* scr = SCR + m * 96;
      if (l < 4) scr[68 + l] = O2[m * kNO + l] + B2l[l];   // meas = hn W^T + b
      kf4(l, PRM + PO_H1, PRM + PO_ER1, X1P + m * 4, P1P + m * 16, scr + 68,
          X1 + m * 4, P1 + m * 16, scr);
    }
    __syncthreads();
  }

  // ---- final carry outputs: x1 then x2, flat (B, SD) each
  if (tid < kMT * 4) {
    int m = tid >> 2, s = tid & 3;
    size_t base = (size_t)kB * kT * kOUT;
    outp[base + (size_t)(row0 + m) * kSD + s] = X1[m * 4 + s];
    outp[base + (size_t)kB * kSD + (size_t)(row0 + m) * kSD + s] = X2[m * 4 + s];
  }
}

extern "C" void kernel_launch(void* const* d_in, const int* in_sizes, int n_in,
                              void* d_out, int out_size, void* d_ws, size_t ws_size,
                              hipStream_t stream) {
  (void)in_sizes; (void)n_in; (void)d_ws; (void)ws_size; (void)out_size;
  const float* inp   = (const float*)d_in[0];
  const float* l2w1  = (const float*)d_in[1];
  const float* l2b1  = (const float*)d_in[2];
  const float* l2w2  = (const float*)d_in[3];
  const float* l2b2  = (const float*)d_in[4];
  const float* l2H   = (const float*)d_in[5];
  const float* l2lQ  = (const float*)d_in[6];
  const float* l2lR  = (const float*)d_in[7];
  const float* l1w1  = (const float*)d_in[8];
  const float* l1b1  = (const float*)d_in[9];
  const float* l1w2  = (const float*)d_in[10];
  const float* l1b2  = (const float*)d_in[11];
  const float* l1H   = (const float*)d_in[12];
  const float* l1lQ  = (const float*)d_in[13];
  const float* l1lR  = (const float*)d_in[14];
  const float* wih   = (const float*)d_in[15];
  const float* whh   = (const float*)d_in[16];
  const float* bih   = (const float*)d_in[17];
  const float* bhh   = (const float*)d_in[18];
  const float* measw = (const float*)d_in[19];
  const float* measb = (const float*)d_in[20];
  const float* outw  = (const float*)d_in[21];
  const float* outb  = (const float*)d_in[22];
  float* outp = (float*)d_out;

  // ~268 KB dynamic LDS per WG (gfx1250 WGP has 320 KB) -> raise the cap.
  (void)hipFuncSetAttribute((const void*)hpr_kernel,
                            hipFuncAttributeMaxDynamicSharedMemorySize, SM_TOTAL);
  hpr_kernel<<<dim3(kB / kMT), dim3(kThreads), SM_TOTAL, stream>>>(
      inp, l2w1, l2b1, l2w2, l2b2, l2H, l2lQ, l2lR,
      l1w1, l1b1, l1w2, l1b2, l1H, l1lQ, l1lR,
      wih, whh, bih, bhh, measw, measb, outw, outb, outp);
}
